// RegDGCNN_23321672417296
// MI455X (gfx1250) — matrix operand order repaired and checked by reading, not compiled, not run on hardware
//
#include <hip/hip_runtime.h>
#include <hip/hip_bf16.h>
#include <stdint.h>

// ---------------------------------------------------------------------------
// RegDGCNN on MI455X (gfx1250): bf16 WMMA for Gram (kNN) + edge MLP GEMMs,
// async global->LDS staging (CDNA5), atomic-max for the neighbor reduction.
// ---------------------------------------------------------------------------

typedef __attribute__((ext_vector_type(16))) __bf16 v16bf;
typedef __attribute__((ext_vector_type(8)))  float  v8f;
typedef __attribute__((ext_vector_type(4)))  int    v4i;

#define NPTS 4096
#define NBATCH 4
#define KNN 20
#define TOTPTS (NBATCH * NPTS)
#define BN_INV 0.99999500003749968750f  // 1/sqrt(1+1e-5)

#define AS1 __attribute__((address_space(1)))
#define AS3 __attribute__((address_space(3)))

#if defined(__has_builtin)
#if __has_builtin(__builtin_amdgcn_global_load_async_to_lds_b128)
#define HAVE_ASYNC 1
#endif
#endif
#ifndef HAVE_ASYNC
#define HAVE_ASYNC 0
#endif

// Copy 16 bytes global -> LDS. Async path uses the CDNA5 async-to-LDS DMA
// (tracked by ASYNCcnt); fallback is a plain b128 load + ds_store_b128.
__device__ __forceinline__ void stage16(const void* g, void* l) {
#if HAVE_ASYNC
  __builtin_amdgcn_global_load_async_to_lds_b128((AS1 v4i*)g, (AS3 v4i*)l, 0, 0);
#else
  *(uint4*)l = *(const uint4*)g;
#endif
}

__device__ __forceinline__ void wait_async0() {
#if HAVE_ASYNC
#if __has_builtin(__builtin_amdgcn_s_wait_asynccnt)
  __builtin_amdgcn_s_wait_asynccnt(0);
#else
  asm volatile("s_wait_asynccnt 0x0" ::: "memory");
#endif
#endif
}

// ---- WMMA operand builders (contiguous k = 16*(lane/16) + s in both A and B) ----------
// A: 16x32 tile, rows striped over lanes 0..15 (lanes 16..31 = K+16 halves).
__device__ __forceinline__ v16bf load_a_tile(const __bf16* src, int rowBase,
                                             int k0, int stride, int lane) {
  int row = rowBase + (lane & 15);
  int hi  = (lane >> 4) & 1;
  const uint32_t* rp = (const uint32_t*)(src + (size_t)row * stride + k0);
  union { v16bf v; uint32_t d[8]; } u;
#pragma unroll
  for (int p = 0; p < 8; ++p) {
    int k = 2 * p + (hi ? 16 : 0);
    u.d[p] = rp[k >> 1];
  }
  return u.v;
}

// B: 32x16 tile; column n contiguous along K (src[col*stride + k]).
__device__ __forceinline__ v16bf load_b_tile(const __bf16* src, int colBase,
                                             int k0, int stride, int lane) {
  int col = colBase + (lane & 15);
  int hi  = (lane >> 4) & 1;
  const uint32_t* cp =
      (const uint32_t*)(src + (size_t)col * stride + k0 + (hi ? 16 : 0));
  union { v16bf v; uint32_t d[8]; } u;
#pragma unroll
  for (int p = 0; p < 8; ++p) u.d[p] = cp[p];
  return u.v;
}

// ---------------------------------------------------------------------------
// Prep / conversion kernels
// ---------------------------------------------------------------------------
__global__ void prep_kernel(const float* __restrict__ pos,
                            const float* __restrict__ nrm,
                            __bf16* __restrict__ xh0) {
  int p = blockIdx.x * blockDim.x + threadIdx.x;
  if (p >= TOTPTS) return;
  __bf16* r = xh0 + (size_t)p * 32;
#pragma unroll
  for (int c = 0; c < 3; ++c) r[c] = (__bf16)pos[p * 3 + c];
#pragma unroll
  for (int c = 0; c < 3; ++c) r[3 + c] = (__bf16)nrm[p * 3 + c];
#pragma unroll
  for (int c = 6; c < 32; ++c) r[c] = (__bf16)0.0f;
}

__global__ void tobf16_kernel(const float* __restrict__ x,
                              __bf16* __restrict__ xh, int n) {
  int i = blockIdx.x * blockDim.x + threadIdx.x;
  if (i < n) xh[i] = (__bf16)x[i];
}

// W (Kdim x Cout, row-major fp32) -> Wt[n][kpad] bf16 (K contiguous per col)
__global__ void wt_kernel(const float* __restrict__ W, __bf16* __restrict__ Wt,
                          int Kdim, int Kpad, int Cout) {
  int i = blockIdx.x * blockDim.x + threadIdx.x;
  if (i >= Kpad * Cout) return;
  int n = i / Kpad, k = i % Kpad;
  Wt[(size_t)n * Kpad + k] =
      (k < Kdim) ? (__bf16)W[(size_t)k * Cout + n] : (__bf16)0.0f;
}

template <int CP>
__global__ void norm_kernel(const __bf16* __restrict__ xh,
                            float* __restrict__ norms) {
  int p = blockIdx.x * blockDim.x + threadIdx.x;
  if (p >= TOTPTS) return;
  const __bf16* r = xh + (size_t)p * CP;
  float s = 0.f;
#pragma unroll 8
  for (int k = 0; k < CP; ++k) {
    float v = (float)r[k];
    s += v * v;
  }
  norms[p] = s;
}

// ---------------------------------------------------------------------------
// kNN: Gram via WMMA; columns staged into LDS (async); per-row top-20.
// Block: 128 threads (4 waves), 16-row panel x 64-column chunks.
// ---------------------------------------------------------------------------
template <int CP>
__global__ void knn_kernel(const __bf16* __restrict__ xh,
                           const float* __restrict__ norms,
                           int* __restrict__ idx) {
  constexpr int CSTR = CP + 8;   // padded LDS row stride (bank-conflict free)
  constexpr int CPR  = CP / 8;   // 16B chunks per feature row
  int rowBase = blockIdx.x * 16;
  int bb      = blockIdx.y * NPTS;
  int tid  = threadIdx.x;
  int lane = tid & 31;
  int wave = tid >> 5;

  __shared__ float distLDS[16][68];
  __shared__ float rn[16];
  __shared__ __align__(16) __bf16 colbuf[64 * CSTR];

  if (tid < 16) rn[tid] = norms[bb + rowBase + tid];

  const __bf16* xb = xh + (size_t)bb * CP;

  // Hoist A operands (loop-invariant row panel) into registers.
  v16bf aReg[CP / 32];
#pragma unroll
  for (int s = 0; s < CP / 32; ++s)
    aReg[s] = load_a_tile(xb, rowBase, s * 32, CP, lane);

  float bestD[KNN];
  int   bestI[KNN];
  float worst = 3.4e38f;
#pragma unroll
  for (int q = 0; q < KNN; ++q) { bestD[q] = 3.4e38f; bestI[q] = bb; }

  for (int cb = 0; cb < NPTS; cb += 64) {
    // stage 64 column feature rows into LDS (async DMA on CDNA5)
    for (int c = tid; c < 64 * CPR; c += 128) {
      int r = c / CPR, cc = c % CPR;
      stage16(xb + (size_t)(cb + r) * CP + cc * 8,
              colbuf + (size_t)r * CSTR + cc * 8);
    }
    wait_async0();
    __syncthreads();

    v8f acc = {};
#pragma unroll
    for (int s = 0; s < CP / 32; ++s) {
      v16bf b = load_b_tile(colbuf, wave * 16, s * 32, CSTR, lane);
      acc = __builtin_amdgcn_wmma_f32_16x16x32_bf16(false, aReg[s], false, b,
                                                    (short)0, acc, false, false);
    }
    int nl = lane & 15;
    int mh = ((lane >> 4) << 3);
    float cnv = norms[bb + cb + wave * 16 + nl];
#pragma unroll
    for (int r = 0; r < 8; ++r) {
      int m = mh + r;
      distLDS[m][wave * 16 + nl] = rn[m] + cnv - 2.0f * acc[r];
    }
    __syncthreads();
    if (tid < 16) {
      for (int j = 0; j < 64; ++j) {
        float d = distLDS[tid][j];
        if (d < worst) {
          int wi = 0; float wv = bestD[0];
#pragma unroll
          for (int q = 1; q < KNN; ++q)
            if (bestD[q] > wv) { wv = bestD[q]; wi = q; }
          bestD[wi] = d; bestI[wi] = bb + cb + j;
          wv = bestD[0];
#pragma unroll
          for (int q = 1; q < KNN; ++q)
            if (bestD[q] > wv) wv = bestD[q];
          worst = wv;
        }
      }
    }
    __syncthreads();
  }
  if (tid < 16) {
    int p = bb + rowBase + tid;
#pragma unroll
    for (int q = 0; q < KNN; ++q) idx[p * KNN + q] = bestI[q];
  }
}

// ---------------------------------------------------------------------------
// EdgeConv: async-stage xi/xj rows -> build E=[xi, xj-xi] in LDS (bf16),
// layer1 WMMA -> LDS hidden, layer2 WMMA, BN+ReLU, atomic-max scatter.
// Block: 128 threads (4 waves), TP points (TP*20 edges, multiple of 16).
// All LDS matrix rows padded by +8 bf16 to avoid bank conflicts.
// ---------------------------------------------------------------------------
template <int CIN, int CPIN, int K1, int COUT, int TP>
__global__ void edge_conv_kernel(
    const __bf16* __restrict__ xh, const int* __restrict__ idx,
    const __bf16* __restrict__ w1t, const float* __restrict__ b1,
    const float* __restrict__ g1, const float* __restrict__ bt1,
    const __bf16* __restrict__ w2t, const float* __restrict__ b2,
    const float* __restrict__ g2, const float* __restrict__ bt2,
    float* __restrict__ xout) {
  constexpr int EDGES = TP * KNN;
  constexpr int MT    = EDGES / 16;
  constexpr int ES    = K1 + 8;     // E row stride
  constexpr int HS    = COUT + 8;   // H row stride
  constexpr int XS    = CPIN + 8;   // staging row stride
  constexpr int CPR   = CPIN / 8;   // 16B chunks per feature row
  static_assert(EDGES % 16 == 0, "edges per block must tile by 16");

  extern __shared__ __align__(16) char smem[];
  __bf16* E  = (__bf16*)smem;                   // [EDGES][ES]
  __bf16* H  = E + (size_t)EDGES * ES;          // [EDGES][HS]
  __bf16* XJ = H + (size_t)EDGES * HS;          // [EDGES][XS]
  __bf16* XI = XJ + (size_t)EDGES * XS;         // [TP][XS]

  int tid  = threadIdx.x;
  int lane = tid & 31;
  int wave = tid >> 5;
  int bb    = blockIdx.y * NPTS;
  int pBase = bb + blockIdx.x * TP;

  // stage neighbor rows (gather) and center rows via async global->LDS
  for (int c = tid; c < EDGES * CPR; c += blockDim.x) {
    int e = c / CPR, cc = c % CPR;
    int p = pBase + e / KNN;
    int j = idx[p * KNN + (e % KNN)];
    stage16(xh + (size_t)j * CPIN + cc * 8, XJ + (size_t)e * XS + cc * 8);
  }
  for (int c = tid; c < TP * CPR; c += blockDim.x) {
    int pl = c / CPR, cc = c % CPR;
    stage16(xh + (size_t)(pBase + pl) * CPIN + cc * 8,
            XI + (size_t)pl * XS + cc * 8);
  }
  wait_async0();
  __syncthreads();

  // build E = [xi, xj - xi] (zero-padded to K1)
  for (int e = tid; e < EDGES; e += blockDim.x) {
    int pl = e / KNN;
    const __bf16* xi = XI + (size_t)pl * XS;
    const __bf16* xj = XJ + (size_t)e * XS;
    __bf16* er = E + (size_t)e * ES;
    for (int k = 0; k < K1; ++k) {
      float v;
      if (k < CIN)          v = (float)xi[k];
      else if (k < 2 * CIN) v = (float)xj[k - CIN] - (float)xi[k - CIN];
      else                  v = 0.0f;
      er[k] = (__bf16)v;
    }
  }
  __syncthreads();

  // layer 1: H = relu(bn(E @ W1 + b1))
  for (int mt = wave; mt < MT; mt += 4) {
    for (int nt = 0; nt < COUT / 16; ++nt) {
      __builtin_prefetch(w1t + (size_t)(nt * 16) * K1, 0, 1);
      v8f acc = {};
#pragma unroll
      for (int s = 0; s < K1 / 32; ++s) {
        v16bf a = load_a_tile(E, mt * 16, s * 32, ES, lane);
        v16bf b = load_b_tile(w1t, nt * 16, s * 32, K1, lane);
        acc = __builtin_amdgcn_wmma_f32_16x16x32_bf16(
            false, a, false, b, (short)0, acc, false, false);
      }
      int n = nt * 16 + (lane & 15);
      float bias = b1[n], sc = g1[n] * BN_INV, bt = bt1[n];
      int mh = ((lane >> 4) << 3);
#pragma unroll
      for (int r = 0; r < 8; ++r) {
        float v = (acc[r] + bias) * sc + bt;
        v = v > 0.f ? v : 0.f;
        H[(size_t)(mt * 16 + mh + r) * HS + n] = (__bf16)v;
      }
    }
  }
  __syncthreads();

  // layer 2 + max-over-K via atomic max (outputs >= 0, xout zero-inited)
  for (int mt = wave; mt < MT; mt += 4) {
    for (int nt = 0; nt < COUT / 16; ++nt) {
      __builtin_prefetch(w2t + (size_t)(nt * 16) * COUT, 0, 1);
      v8f acc = {};
#pragma unroll
      for (int s = 0; s < COUT / 32; ++s) {
        v16bf a = load_a_tile(H, mt * 16, s * 32, HS, lane);
        v16bf b = load_b_tile(w2t, nt * 16, s * 32, COUT, lane);
        acc = __builtin_amdgcn_wmma_f32_16x16x32_bf16(
            false, a, false, b, (short)0, acc, false, false);
      }
      int n = nt * 16 + (lane & 15);
      float bias = b2[n], sc = g2[n] * BN_INV, bt = bt2[n];
      int mh = ((lane >> 4) << 3);
#pragma unroll
      for (int r = 0; r < 8; ++r) {
        int m = mt * 16 + mh + r;
        float v = (acc[r] + bias) * sc + bt;
        v = v > 0.f ? v : 0.f;
        int p = pBase + m / KNN;
        atomicMax((int*)(xout + (size_t)p * COUT + n), __float_as_int(v));
      }
    }
  }
}

// ---------------------------------------------------------------------------
// Mean pool into feat[B][960]
// ---------------------------------------------------------------------------
__global__ void mean_kernel(const float* __restrict__ x, int C,
                            float* __restrict__ feat, int off) {
  int i = blockIdx.x * blockDim.x + threadIdx.x;
  if (i >= NBATCH * C) return;
  int b = i / C, c = i % C;
  const float* xp = x + ((size_t)b * NPTS) * C + c;
  float s = 0.f;
  for (int n = 0; n < NPTS; ++n) s += xp[(size_t)n * C];
  feat[b * 960 + off + c] = s * (1.0f / NPTS);
}

// ---------------------------------------------------------------------------
// Head: feat(4x960) -> 512 -> 256 -> 1
// ---------------------------------------------------------------------------
__global__ void head_kernel(const float* __restrict__ feat,
                            const float* __restrict__ W1, const float* __restrict__ b1,
                            const float* __restrict__ g1, const float* __restrict__ bt1,
                            const float* __restrict__ W2, const float* __restrict__ b2,
                            const float* __restrict__ g2, const float* __restrict__ bt2,
                            const float* __restrict__ W3, const float* __restrict__ b3,
                            float* __restrict__ out) {
  __shared__ float h1[NBATCH * 512];
  __shared__ float h2[NBATCH * 256];
  int tid = threadIdx.x;
  for (int t = tid; t < NBATCH * 512; t += blockDim.x) {
    int b = t / 512, c = t % 512;
    float s = 0.f;
    for (int k = 0; k < 960; ++k) s += feat[b * 960 + k] * W1[k * 512 + c];
    s = (s + b1[c]) * (g1[c] * BN_INV) + bt1[c];
    h1[t] = s > 0.f ? s : 0.f;
  }
  __syncthreads();
  for (int t = tid; t < NBATCH * 256; t += blockDim.x) {
    int b = t / 256, c = t % 256;
    float s = 0.f;
    for (int k = 0; k < 512; ++k) s += h1[b * 512 + k] * W2[k * 256 + c];
    s = (s + b2[c]) * (g2[c] * BN_INV) + bt2[c];
    h2[t] = s > 0.f ? s : 0.f;
  }
  __syncthreads();
  if (tid < NBATCH) {
    float s = 0.f;
    for (int k = 0; k < 256; ++k) s += h2[tid * 256 + k] * W3[k];
    out[tid] = s + b3[0];
  }
}

// ---------------------------------------------------------------------------
// Launch
// ---------------------------------------------------------------------------
extern "C" void kernel_launch(void* const* d_in, const int* in_sizes, int n_in,
                              void* d_out, int out_size, void* d_ws,
                              size_t ws_size, hipStream_t stream) {
  (void)in_sizes; (void)n_in; (void)out_size; (void)ws_size;
  auto F = [&](int i) { return (const float*)d_in[i]; };
  const float* pos = F(0);
  const float* nrm = F(1);
  const int c1 = 3, c2 = 11, c3 = 19, c4 = 27;
  const float* l1W = F(35); const float* l1b = F(36);
  const float* n1g = F(37); const float* n1b = F(38);
  const float* l2W = F(39); const float* l2b = F(40);
  const float* n2g = F(41); const float* n2b = F(42);
  const float* l3W = F(43); const float* l3b = F(44);

  char* ws = (char*)d_ws;
  size_t off = 0;
  auto alloc = [&](size_t bytes) -> void* {
    void* p = ws + off;
    off = (off + bytes + 255) & ~(size_t)255;
    return p;
  };
  __bf16* xh0 = (__bf16*)alloc((size_t)TOTPTS * 32 * 2);
  __bf16* xh1 = (__bf16*)alloc((size_t)TOTPTS * 64 * 2);
  __bf16* xh2 = (__bf16*)alloc((size_t)TOTPTS * 128 * 2);
  __bf16* xh3 = (__bf16*)alloc((size_t)TOTPTS * 256 * 2);
  float* x1f = (float*)alloc((size_t)TOTPTS * 64 * 4);
  float* x2f = (float*)alloc((size_t)TOTPTS * 128 * 4);
  float* x3f = (float*)alloc((size_t)TOTPTS * 256 * 4);
  float* x4f = (float*)alloc((size_t)TOTPTS * 512 * 4);
  float* norms = (float*)alloc((size_t)TOTPTS * 4);
  int*   idx   = (int*)alloc((size_t)TOTPTS * KNN * 4);
  __bf16* w1t1 = (__bf16*)alloc((size_t)64 * 32 * 2);
  __bf16* w2t1 = (__bf16*)alloc((size_t)64 * 64 * 2);
  __bf16* w1t2 = (__bf16*)alloc((size_t)128 * 128 * 2);
  __bf16* w2t2 = (__bf16*)alloc((size_t)128 * 128 * 2);
  __bf16* w1t3 = (__bf16*)alloc((size_t)256 * 256 * 2);
  __bf16* w2t3 = (__bf16*)alloc((size_t)256 * 256 * 2);
  __bf16* w1t4 = (__bf16*)alloc((size_t)512 * 512 * 2);
  __bf16* w2t4 = (__bf16*)alloc((size_t)512 * 512 * 2);
  float* feat = (float*)alloc((size_t)NBATCH * 960 * 4);

  (void)hipMemsetAsync(x1f, 0, (size_t)TOTPTS * 64 * 4, stream);
  (void)hipMemsetAsync(x2f, 0, (size_t)TOTPTS * 128 * 4, stream);
  (void)hipMemsetAsync(x3f, 0, (size_t)TOTPTS * 256 * 4, stream);
  (void)hipMemsetAsync(x4f, 0, (size_t)TOTPTS * 512 * 4, stream);

  auto wt = [&](const float* W, __bf16* Wt, int Kd, int Kp, int Co) {
    int n = Kp * Co;
    wt_kernel<<<(n + 255) / 256, 256, 0, stream>>>(W, Wt, Kd, Kp, Co);
  };
  wt(F(c1 + 0), w1t1, 12, 32, 64);
  wt(F(c1 + 4), w2t1, 64, 64, 64);
  wt(F(c2 + 0), w1t2, 128, 128, 128);
  wt(F(c2 + 4), w2t2, 128, 128, 128);
  wt(F(c3 + 0), w1t3, 256, 256, 256);
  wt(F(c3 + 4), w2t3, 256, 256, 256);
  wt(F(c4 + 0), w1t4, 512, 512, 512);
  wt(F(c4 + 4), w2t4, 512, 512, 512);

  prep_kernel<<<(TOTPTS + 255) / 256, 256, 0, stream>>>(pos, nrm, xh0);

  dim3 knnGrid(NPTS / 16, NBATCH);
  // edge-conv dynamic LDS: rows padded by +8 bf16 everywhere
  auto edge_sh = [](int TP, int K1, int CO, int CPIN) -> size_t {
    int EDGES = TP * KNN;
    return ((size_t)EDGES * (K1 + 8) + (size_t)EDGES * (CO + 8) +
            (size_t)EDGES * (CPIN + 8) + (size_t)TP * (CPIN + 8)) * 2;
  };

  // ---- stage 1 ----
  norm_kernel<32><<<(TOTPTS + 255) / 256, 256, 0, stream>>>(xh0, norms);
  knn_kernel<32><<<knnGrid, 128, 0, stream>>>(xh0, norms, idx);
  edge_conv_kernel<6, 32, 32, 64, 16>
      <<<dim3(NPTS / 16, NBATCH), 128, edge_sh(16, 32, 64, 32), stream>>>(
          xh0, idx, w1t1, F(c1 + 1), F(c1 + 2), F(c1 + 3), w2t1, F(c1 + 5),
          F(c1 + 6), F(c1 + 7), x1f);
  tobf16_kernel<<<(TOTPTS * 64 + 255) / 256, 256, 0, stream>>>(x1f, xh1,
                                                               TOTPTS * 64);

  // ---- stage 2 ----
  norm_kernel<64><<<(TOTPTS + 255) / 256, 256, 0, stream>>>(xh1, norms);
  knn_kernel<64><<<knnGrid, 128, 0, stream>>>(xh1, norms, idx);
  edge_conv_kernel<64, 64, 128, 128, 8>
      <<<dim3(NPTS / 8, NBATCH), 128, edge_sh(8, 128, 128, 64), stream>>>(
          xh1, idx, w1t2, F(c2 + 1), F(c2 + 2), F(c2 + 3), w2t2, F(c2 + 5),
          F(c2 + 6), F(c2 + 7), x2f);
  tobf16_kernel<<<(TOTPTS * 128 + 255) / 256, 256, 0, stream>>>(x2f, xh2,
                                                                TOTPTS * 128);

  // ---- stage 3 ----
  norm_kernel<128><<<(TOTPTS + 255) / 256, 256, 0, stream>>>(xh2, norms);
  knn_kernel<128><<<knnGrid, 128, 0, stream>>>(xh2, norms, idx);
  edge_conv_kernel<128, 128, 256, 256, 8>
      <<<dim3(NPTS / 8, NBATCH), 128, edge_sh(8, 256, 256, 128), stream>>>(
          xh2, idx, w1t3, F(c3 + 1), F(c3 + 2), F(c3 + 3), w2t3, F(c3 + 5),
          F(c3 + 6), F(c3 + 7), x3f);
  tobf16_kernel<<<(TOTPTS * 256 + 255) / 256, 256, 0, stream>>>(x3f, xh3,
                                                                TOTPTS * 256);

  // ---- stage 4 ----
  norm_kernel<256><<<(TOTPTS + 255) / 256, 256, 0, stream>>>(xh3, norms);
  knn_kernel<256><<<knnGrid, 128, 0, stream>>>(xh3, norms, idx);
  edge_conv_kernel<256, 256, 512, 512, 4>
      <<<dim3(NPTS / 4, NBATCH), 128, edge_sh(4, 512, 512, 256), stream>>>(
          xh3, idx, w1t4, F(c4 + 1), F(c4 + 2), F(c4 + 3), w2t4, F(c4 + 5),
          F(c4 + 6), F(c4 + 7), x4f);

  // ---- mean pool + head ----
  mean_kernel<<<(NBATCH * 64 + 255) / 256, 256, 0, stream>>>(x1f, 64, feat, 0);
  mean_kernel<<<(NBATCH * 128 + 255) / 256, 256, 0, stream>>>(x2f, 128, feat, 64);
  mean_kernel<<<(NBATCH * 256 + 255) / 256, 256, 0, stream>>>(x3f, 256, feat, 192);
  mean_kernel<<<(NBATCH * 512 + 255) / 256, 256, 0, stream>>>(x4f, 512, feat, 448);

  head_kernel<<<1, 512, 0, stream>>>(feat, l1W, l1b, n1g, n1b, l2W, l2b, n2g,
                                     n2b, l3W, l3b, (float*)d_out);
}
